// LSTM_RFCA_46995532153065
// MI455X (gfx1250) — compile-verified, matrix operand order checked by reading, not run on hardware
//
#include <hip/hip_runtime.h>
#include <math.h>

// ---------------------------------------------------------------------------
// MI455X / gfx1250, wave32, WMMA f32<=f16.
// Pipeline: gcn -> bn1/relu -> rfca -> gat -> rfca -> 4x lstm-gates -> head.
// Weights + activations pre-converted to f16 once per launch so the WMMA
// inner loops do no f32->f16 conversion and read half-width operands.
// A-tiles staged to LDS via GLOBAL_LOAD_ASYNC_TO_LDS_B128 when available.
// ---------------------------------------------------------------------------

#define DF   128        // feature dim
#define KRF  1152       // rfca GEMM K = D*9
#define EPSV 1e-5f

typedef __attribute__((ext_vector_type(16))) _Float16 v16h;
typedef __attribute__((ext_vector_type(8)))  float    v8f;
typedef __attribute__((__vector_size__(16))) int      ai4;   // async b128 payload

#if defined(__has_builtin)
#if __has_builtin(__builtin_amdgcn_global_load_async_to_lds_b128) && \
    __has_builtin(__builtin_amdgcn_s_wait_asynccnt)
#define USE_ASYNC_COPY 1
#endif
#endif
#ifndef USE_ASYNC_COPY
#define USE_ASYNC_COPY 0
#endif

static __device__ __forceinline__ float sigmoidf_(float x) {
  return 1.f / (1.f + __expf(-x));
}

// K-offset within a 16x32 f16 A/B fragment for element e of `lane`
// (per CDNA5 ISA 7.12.2 16-bit A layout; B symmetric, K-major).
static __device__ __forceinline__ int frag_k(int lane, int e) {
  int khalf = (lane < 16) ? 0 : 8;
  return (e < 8) ? (khalf + e) : (khalf + 8 + e);   // e>=8 -> khalf+16+(e-8)
}

// monotonic float<->uint encoding for atomicMax on floats
static __device__ __forceinline__ unsigned fenc(float f) {
  unsigned u = __float_as_uint(f);
  return (u & 0x80000000u) ? ~u : (u | 0x80000000u);
}
static __device__ __forceinline__ float fdec(unsigned k) {
  unsigned u = (k & 0x80000000u) ? (k & 0x7FFFFFFFu) : ~k;
  return __uint_as_float(u);
}

// Stage one 16x128 f16 tile into LDS: 256 threads x 16B each.
static __device__ __forceinline__ void stage_tile_f16(
    const _Float16* __restrict__ A16, _Float16 (*sA)[DF],
    int nodeBase, int N, int tid) {
  int r  = tid >> 4;          // 0..15 row
  int c8 = (tid & 15) * 8;    // halves column base
  _Float16* lp = &sA[r][c8];
  int n = nodeBase + r;
#if USE_ASYNC_COPY
  if (n < N) {
    __builtin_amdgcn_global_load_async_to_lds_b128(
        (ai4*)(A16 + (size_t)n * DF + c8), (ai4*)lp, 0, 0);
  } else {
#pragma unroll
    for (int j = 0; j < 8; ++j) lp[j] = (_Float16)0.f;
  }
  __builtin_amdgcn_s_wait_asynccnt(0);
#else
  if (n < N) {
    *(uint4*)lp = *(const uint4*)(A16 + (size_t)n * DF + c8);
  } else {
#pragma unroll
    for (int j = 0; j < 8; ++j) lp[j] = (_Float16)0.f;
  }
#endif
}

// ------------------------------ elementwise --------------------------------
__global__ __launch_bounds__(256) void zero_f(float* p, int n) {
  int i = blockIdx.x * 256 + threadIdx.x;
  if (i < n) p[i] = 0.f;
}
__global__ __launch_bounds__(256) void cvt_f16(const float* __restrict__ in,
                                               _Float16* __restrict__ out,
                                               int n) {
  int i = blockIdx.x * 256 + threadIdx.x;
  if (i < n) out[i] = (_Float16)in[i];
}
__global__ __launch_bounds__(256) void init_deg(float* deg, int n) {
  int i = blockIdx.x * 256 + threadIdx.x;
  if (i < n) deg[i] = 1.f;                 // self loop contributes 1
}
__global__ __launch_bounds__(256) void accum_deg(const int* __restrict__ ei,
                                                 float* deg, int E) {
  int e = blockIdx.x * 256 + threadIdx.x;
  if (e < E) atomicAdd(&deg[ei[e]], 1.f);  // row = ei[0][e]
}
__global__ __launch_bounds__(256) void deg_to_dinv(float* deg, int n) {
  int i = blockIdx.x * 256 + threadIdx.x;
  if (i < n) deg[i] = rsqrtf(deg[i]);
}
// out = relu(bn(in + addb)) per 128-channel
__global__ __launch_bounds__(256) void add_bn_relu(
    const float* __restrict__ in, const float* __restrict__ addb,
    const float* __restrict__ g, const float* __restrict__ b,
    const float* __restrict__ m, const float* __restrict__ v,
    float* __restrict__ out, int total) {
  int i = blockIdx.x * 256 + threadIdx.x;
  if (i >= total) return;
  int c = i & (DF - 1);
  float x = in[i] + (addb ? addb[c] : 0.f);
  x = (x - m[c]) * g[c] * rsqrtf(v[c] + EPSV) + b[c];
  out[i] = fmaxf(x, 0.f);
}

// ------------------------ generic 128x128 WMMA GEMM ------------------------
// out[n,c] = sum_k A16[n,k] * W16[c,k] (+ bias[c]).  block=256 (8 waves),
// 16 nodes per block, one 16-col tile per wave, K=128 in 4 wmma steps.
__global__ __launch_bounds__(256) void gemm128_wmma_f16(
    const _Float16* __restrict__ A16, const _Float16* __restrict__ W16,
    const float* __restrict__ bias, float* __restrict__ out, int N) {
  __shared__ __align__(16) _Float16 sA[16][DF];
  int nodeBase = blockIdx.x * 16;
  int tid = threadIdx.x;
  stage_tile_f16(A16, sA, nodeBase, N, tid);
  __syncthreads();
  int wave = tid >> 5, lane = tid & 31;
  int colBase = wave * 16;
  int mrow = lane & 15;
  v8f acc = {};
  for (int k0 = 0; k0 < DF; k0 += 32) {
    v16h a, b;
#pragma unroll
    for (int e = 0; e < 16; ++e) {
      int kk = frag_k(lane, e);
      a[e] = sA[mrow][k0 + kk];
      b[e] = W16[(size_t)(colBase + mrow) * DF + k0 + kk];
    }
    if (k0 + 32 < DF)   // pull next weight slice toward L2 (global_prefetch_b8)
      __builtin_prefetch(&W16[(size_t)(colBase + mrow) * DF + k0 + 32], 0, 1);
    acc = __builtin_amdgcn_wmma_f32_16x16x32_f16(false, a, false, b,
                                                 (short)0, acc, false, false);
  }
#pragma unroll
  for (int r = 0; r < 8; ++r) {
    int mm = r + ((lane < 16) ? 0 : 8);
    int n = nodeBase + mm;
    if (n >= N) continue;
    int c = colBase + (lane & 15);
    float vv = acc[r];
    if (bias) vv += bias[c];
    out[(size_t)n * DF + c] = vv;
  }
}

// ------------------------------- GCN scatter -------------------------------
__global__ __launch_bounds__(128) void gcn_scatter(
    const int* __restrict__ ei, const float* __restrict__ dinv,
    const float* __restrict__ zw, float* __restrict__ agg, int E, int N) {
  int e = blockIdx.x;           // E edges + N self loops
  int r, c;
  if (e < E) { r = ei[e]; c = ei[E + e]; } else { r = c = e - E; }
  float coef = dinv[r] * dinv[c];
  int f = threadIdx.x;
  atomicAdd(&agg[(size_t)c * DF + f], coef * zw[(size_t)r * DF + f]);
}

// ------------------------------- GAT ---------------------------------------
__global__ __launch_bounds__(256) void gat_coef(
    const float* __restrict__ zw, const float* __restrict__ asrc,
    const float* __restrict__ adst, float* zs, float* zd, int N) {
  int n = blockIdx.x * 256 + threadIdx.x;
  if (n >= N) return;
  float ss = 0.f, sd = 0.f;
  for (int k = 0; k < DF; ++k) {
    float z = zw[(size_t)n * DF + k];
    ss += z * asrc[k];
    sd += z * adst[k];
  }
  zs[n] = ss; zd[n] = sd;
}
__global__ __launch_bounds__(256) void gat_max(
    const int* __restrict__ ei, const float* __restrict__ zs,
    const float* __restrict__ zd, unsigned* mEnc, int E, int N) {
  int t = blockIdx.x * 256 + threadIdx.x;
  if (t >= E + N) return;
  int r, c;
  if (t < E) { r = ei[t]; c = ei[E + t]; } else { r = c = t - E; }
  float ev = zs[r] + zd[c];
  ev = (ev < 0.f) ? 0.2f * ev : ev;        // leaky_relu
  atomicMax(&mEnc[c], fenc(ev));
}
__global__ __launch_bounds__(256) void gat_sum(
    const int* __restrict__ ei, const float* __restrict__ zs,
    const float* __restrict__ zd, const unsigned* __restrict__ mEnc,
    float* sSum, int E, int N) {
  int t = blockIdx.x * 256 + threadIdx.x;
  if (t >= E + N) return;
  int r, c;
  if (t < E) { r = ei[t]; c = ei[E + t]; } else { r = c = t - E; }
  float ev = zs[r] + zd[c];
  ev = (ev < 0.f) ? 0.2f * ev : ev;
  atomicAdd(&sSum[c], __expf(ev - fdec(mEnc[c])));
}
__global__ __launch_bounds__(128) void gat_scatter(
    const int* __restrict__ ei, const float* __restrict__ zs,
    const float* __restrict__ zd, const unsigned* __restrict__ mEnc,
    const float* __restrict__ sSum, const float* __restrict__ zw,
    float* __restrict__ agg, int E, int N) {
  int e = blockIdx.x;
  int r, c;
  if (e < E) { r = ei[e]; c = ei[E + e]; } else { r = c = e - E; }
  float ev = zs[r] + zd[c];
  ev = (ev < 0.f) ? 0.2f * ev : ev;
  float alpha = __expf(ev - fdec(mEnc[c])) / sSum[c];
  int f = threadIdx.x;
  atomicAdd(&agg[(size_t)c * DF + f], alpha * zw[(size_t)r * DF + f]);
}

// ------------------------------- RFCA (fused) ------------------------------
static __device__ __forceinline__ void gen_row(
    float hval, int d, const float* __restrict__ gw,
    const float* __restrict__ gg, const float* __restrict__ gb,
    const float* __restrict__ gm, const float* __restrict__ gv, float gf[9]) {
#pragma unroll
  for (int j = 0; j < 9; ++j) {
    int p = d * 9 + j;
    float t = hval * gw[p];
    t = (t - gm[p]) * gg[p] * rsqrtf(gv[p] + EPSV) + gb[p];
    gf[j] = fmaxf(t, 0.f);
  }
}

// block = 256 (8 waves) = 16 nodes.  Per half-wave: one node, 8 d's per lane.
// Phase 1: gen + pooled y (shfl reduce) + bnm/hardswish -> sY.
// Phase 2: channel gates -> gated g matrix in LDS (f16, 16x1152).
// Phase 3: WMMA GEMM vs f16 cf_w (36 k-steps) with bn2+relu epilogue;
//          writes f32 h and its f16 mirror for the next GEMM consumer.
__global__ __launch_bounds__(256) void rfca_fused(
    const float* __restrict__ H,
    const float* __restrict__ genw, const float* __restrict__ geng,
    const float* __restrict__ genb, const float* __restrict__ genm,
    const float* __restrict__ genv,
    const float* __restrict__ c1w, const float* __restrict__ c1b,
    const float* __restrict__ bnmg, const float* __restrict__ bnmb,
    const float* __restrict__ bnmm, const float* __restrict__ bnmv,
    const float* __restrict__ chw, const float* __restrict__ chb,
    const float* __restrict__ cww, const float* __restrict__ cwb,
    const _Float16* __restrict__ cfw16, const float* __restrict__ cfb,
    const float* __restrict__ bn2g, const float* __restrict__ bn2b,
    const float* __restrict__ bn2m, const float* __restrict__ bn2v,
    float* __restrict__ out, _Float16* __restrict__ out16, int N) {
  __shared__ __align__(16) _Float16 sG[16][KRF];   // gated g (36 KB)
  __shared__ float sY[16][48];                     // y[m=0..7][h'=0..5]

  int nodeBase = blockIdx.x * 16;
  int tid = threadIdx.x;
  int wave = tid >> 5, lane = tid & 31;
  int node = wave * 2 + (lane >> 4);       // node within tile (0..15)
  int gn = nodeBase + node;
  int dl = lane & 15;

  // ---- phase 1: per-(node,d) gen + row/col means, partial y over d ----
  float acc[48];
#pragma unroll
  for (int i = 0; i < 48; ++i) acc[i] = 0.f;
#pragma unroll
  for (int dd = 0; dd < 8; ++dd) {
    int d = dl + 16 * dd;
    float hv = (gn < N) ? H[(size_t)gn * DF + d] : 0.f;
    float gf[9];
    gen_row(hv, d, genw, geng, genb, genm, genv, gf);
    float yv[6];
#pragma unroll
    for (int i = 0; i < 3; ++i)
      yv[i] = (gf[3 * i] + gf[3 * i + 1] + gf[3 * i + 2]) * (1.f / 3.f); // xh
#pragma unroll
    for (int j = 0; j < 3; ++j)
      yv[3 + j] = (gf[j] + gf[3 + j] + gf[6 + j]) * (1.f / 3.f);         // xw
#pragma unroll
    for (int m = 0; m < 8; ++m) {
      float w = c1w[m * DF + d];
#pragma unroll
      for (int t = 0; t < 6; ++t) acc[m * 6 + t] += w * yv[t];
    }
  }
  // reduce across the 16 lanes of this node (xor 1,2,4,8 stays in half-wave)
#pragma unroll
  for (int off = 8; off >= 1; off >>= 1) {
#pragma unroll
    for (int i = 0; i < 48; ++i) acc[i] += __shfl_xor(acc[i], off, 32);
  }
  if (dl < 8) {
    int m = dl;
#pragma unroll
    for (int t = 0; t < 6; ++t) {
      float y = acc[m * 6 + t] + c1b[m];
      y = (y - bnmm[m]) * bnmg[m] * rsqrtf(bnmv[m] + EPSV) + bnmb[m];
      y = y * fminf(fmaxf(y + 3.f, 0.f), 6.f) * (1.f / 6.f);  // hardswish
      sY[node][m * 6 + t] = y;
    }
  }
  __syncthreads();

  // ---- phase 2: channel attention gates, build gated g in LDS (f16) ----
#pragma unroll
  for (int dd = 0; dd < 8; ++dd) {
    int d = dl + 16 * dd;
    float hv = (gn < N) ? H[(size_t)gn * DF + d] : 0.f;
    float gf[9];
    gen_row(hv, d, genw, geng, genb, genm, genv, gf);
    float ah[3], aw[3];
#pragma unroll
    for (int i = 0; i < 3; ++i) {
      float s = chb[d];
#pragma unroll
      for (int m = 0; m < 8; ++m) s += chw[d * 8 + m] * sY[node][m * 6 + i];
      ah[i] = sigmoidf_(s);
    }
#pragma unroll
    for (int j = 0; j < 3; ++j) {
      float s = cwb[d];
#pragma unroll
      for (int m = 0; m < 8; ++m) s += cww[d * 8 + m] * sY[node][m * 6 + 3 + j];
      aw[j] = sigmoidf_(s);
    }
#pragma unroll
    for (int i = 0; i < 3; ++i)
#pragma unroll
      for (int j = 0; j < 3; ++j)
        sG[node][d * 9 + i * 3 + j] =
            (_Float16)(gf[i * 3 + j] * aw[j] * ah[i]);
  }
  __syncthreads();

  // ---- phase 3: out[16 x 128] = g(16 x 1152) @ cf_w^T, bn2+relu fused ----
  int colBase = wave * 16;
  int mrow = lane & 15;
  v8f acc2 = {};
  for (int k0 = 0; k0 < KRF; k0 += 32) {
    v16h a, b;
#pragma unroll
    for (int e = 0; e < 16; ++e) {
      int kk = frag_k(lane, e);
      a[e] = sG[mrow][k0 + kk];
      b[e] = cfw16[(size_t)(colBase + mrow) * KRF + k0 + kk];
    }
    acc2 = __builtin_amdgcn_wmma_f32_16x16x32_f16(false, a, false, b,
                                                  (short)0, acc2, false, false);
  }
#pragma unroll
  for (int r = 0; r < 8; ++r) {
    int mm = r + ((lane < 16) ? 0 : 8);
    int n = nodeBase + mm;
    if (n >= N) continue;
    int c = colBase + (lane & 15);
    float vv = acc2[r] + cfb[c];
    vv = (vv - bn2m[c]) * bn2g[c] * rsqrtf(bn2v[c] + EPSV) + bn2b[c];
    vv = fmaxf(vv, 0.f);
    out[(size_t)n * DF + c] = vv;
    out16[(size_t)n * DF + c] = (_Float16)vv;
  }
}

// --------------------------- LSTM-gate layer -------------------------------
// gates = H @ wih^T + bih + bhh ; c = sig(i)*tanh(g) ; h = sig(o)*tanh(c)
// (whh and the f-gate are dead in the reference -> skip 128 of 512 cols)
__global__ __launch_bounds__(256) void lstm_layer_wmma(
    const _Float16* __restrict__ H16, const _Float16* __restrict__ wih16,
    const float* __restrict__ bih, const float* __restrict__ bhh,
    float* __restrict__ Hout, _Float16* __restrict__ Hout16, int N) {
  __shared__ __align__(16) _Float16 sA[16][DF];
  int nodeBase = blockIdx.x * 16;
  int tid = threadIdx.x;
  stage_tile_f16(H16, sA, nodeBase, N, tid);
  __syncthreads();
  int wave = tid >> 5, lane = tid & 31;
  int cbase = wave * 16;
  int mrow = lane & 15;
  v8f accI = {}, accG = {}, accO = {};
  for (int k0 = 0; k0 < DF; k0 += 32) {
    v16h a, bI, bG, bO;
#pragma unroll
    for (int e = 0; e < 16; ++e) {
      int kk = frag_k(lane, e);
      a[e] = sA[mrow][k0 + kk];
      const _Float16* wr = wih16 + (size_t)(cbase + mrow) * DF + k0 + kk;
      bI[e] = wr[0];                    // i-gate rows [0,128)
      bG[e] = wr[(size_t)256 * DF];     // g-gate rows [256,384)
      bO[e] = wr[(size_t)384 * DF];     // o-gate rows [384,512)
    }
    accI = __builtin_amdgcn_wmma_f32_16x16x32_f16(false, a, false, bI,
                                                  (short)0, accI, false, false);
    accG = __builtin_amdgcn_wmma_f32_16x16x32_f16(false, a, false, bG,
                                                  (short)0, accG, false, false);
    accO = __builtin_amdgcn_wmma_f32_16x16x32_f16(false, a, false, bO,
                                                  (short)0, accO, false, false);
  }
#pragma unroll
  for (int r = 0; r < 8; ++r) {
    int mm = r + ((lane < 16) ? 0 : 8);
    int n = nodeBase + mm;
    if (n >= N) continue;
    int c = cbase + (lane & 15);
    float iv = accI[r] + bih[c] + bhh[c];
    float gv = accG[r] + bih[256 + c] + bhh[256 + c];
    float ov = accO[r] + bih[384 + c] + bhh[384 + c];
    float cc = sigmoidf_(iv) * tanhf(gv);
    float hv = sigmoidf_(ov) * tanhf(cc);
    Hout[(size_t)n * DF + c] = hv;
    Hout16[(size_t)n * DF + c] = (_Float16)hv;
  }
}

// ------------------------------- head --------------------------------------
__global__ __launch_bounds__(256) void head_kernel(
    const float* __restrict__ H, const float* __restrict__ l1w,
    const float* __restrict__ l1b, const float* __restrict__ l2w,
    const float* __restrict__ l2b, float* __restrict__ out, int N) {
  int n = blockIdx.x * 256 + threadIdx.x;
  if (n >= N) return;
  float h1[64];
  for (int o = 0; o < 64; ++o) {
    float s = l1b[o];
    for (int k = 0; k < DF; ++k) s += H[(size_t)n * DF + k] * l1w[o * DF + k];
    h1[o] = fmaxf(s, 0.f);
  }
  float lg[10], mx = -1e30f;
  for (int o = 0; o < 10; ++o) {
    float s = l2b[o];
    for (int k = 0; k < 64; ++k) s += h1[k] * l2w[o * 64 + k];
    lg[o] = s;
    mx = fmaxf(mx, s);
  }
  float se = 0.f;
  for (int o = 0; o < 10; ++o) se += __expf(lg[o] - mx);
  float lse = mx + __logf(se);
  for (int o = 0; o < 10; ++o) out[(size_t)n * 10 + o] = lg[o] - lse;
}

// ---------------------------------------------------------------------------
extern "C" void kernel_launch(void* const* d_in, const int* in_sizes, int n_in,
                              void* d_out, int out_size, void* d_ws,
                              size_t ws_size, hipStream_t stream) {
  const float* x      = (const float*)d_in[0];
  const int*   ei     = (const int*)d_in[1];
  const float* lin_w  = (const float*)d_in[2];
  const float* lin_b  = (const float*)d_in[3];
  const float* bn1g   = (const float*)d_in[4];
  const float* bn1b   = (const float*)d_in[5];
  const float* bn1m   = (const float*)d_in[6];
  const float* bn1v   = (const float*)d_in[7];
  const float* bn2g   = (const float*)d_in[8];
  const float* bn2b   = (const float*)d_in[9];
  const float* bn2m   = (const float*)d_in[10];
  const float* bn2v   = (const float*)d_in[11];
  const float* genw   = (const float*)d_in[12];
  const float* geng   = (const float*)d_in[13];
  const float* genb   = (const float*)d_in[14];
  const float* genm   = (const float*)d_in[15];
  const float* genv   = (const float*)d_in[16];
  const float* c1w    = (const float*)d_in[17];
  const float* c1b    = (const float*)d_in[18];
  const float* bnmg   = (const float*)d_in[19];
  const float* bnmb   = (const float*)d_in[20];
  const float* bnmm   = (const float*)d_in[21];
  const float* bnmv   = (const float*)d_in[22];
  const float* chw    = (const float*)d_in[23];
  const float* chb    = (const float*)d_in[24];
  const float* cww    = (const float*)d_in[25];
  const float* cwb    = (const float*)d_in[26];
  const float* cfw    = (const float*)d_in[27];
  const float* cfb    = (const float*)d_in[28];
  const float* gatw   = (const float*)d_in[29];
  const float* gasrc  = (const float*)d_in[30];
  const float* gadst  = (const float*)d_in[31];
  const float* gatb   = (const float*)d_in[32];
  const float* wih    = (const float*)d_in[33];
  /* d_in[34] = lstm_whh: dead in the reference (no recurrent term) */
  const float* bih    = (const float*)d_in[35];
  const float* bhh    = (const float*)d_in[36];
  const float* l1w    = (const float*)d_in[37];
  const float* l1b    = (const float*)d_in[38];
  const float* l2w    = (const float*)d_in[39];
  const float* l2b    = (const float*)d_in[40];
  (void)n_in; (void)out_size; (void)ws_size;

  const int N = in_sizes[0] / DF;          // 50000
  const int E = in_sizes[1] / 2;           // 600000
  const size_t nd = (size_t)N * DF;
  const int szLin = DF * DF;               // 16384
  const int szWih = 4 * 512 * DF;          // 262144
  const int szCfw = DF * KRF;              // 147456

  float* ws   = (float*)d_ws;
  float* zw   = ws;                        // N x 128
  float* agg  = zw + nd;                   // N x 128 scatter accumulator
  float* h0   = agg + nd;                  // N x 128
  float* h1   = h0 + nd;                   // N x 128
  float* dinv = h1 + nd;                   // N
  float* zs   = dinv + N;                  // N
  float* zd   = zs + N;                    // N
  unsigned* mEnc = (unsigned*)(zd + N);    // N (ordered-uint encoded max)
  float* sSum = (float*)(mEnc + N);        // N
  uintptr_t p16 = ((uintptr_t)(sSum + N) + 15) & ~(uintptr_t)15;
  _Float16* hA16  = (_Float16*)p16;        // N x 128 f16 activations (ping)
  _Float16* hB16  = hA16 + nd;             // N x 128 f16 activations (pong)
  _Float16* lin16 = hB16 + nd;             // f16 weights
  _Float16* gat16 = lin16 + szLin;
  _Float16* wih16 = gat16 + szLin;
  _Float16* cfw16 = wih16 + szWih;
  float* out  = (float*)d_out;             // N x 10

  const int TB = 256;
  const int gN    = (N + TB - 1) / TB;
  const int gE    = (E + TB - 1) / TB;
  const int gEN   = (E + N + TB - 1) / TB;
  const int gND   = (int)((nd + TB - 1) / TB);
  const int gTile = (N + 15) / 16;

  // --- one-shot f16 conversion of WMMA operands (L2-resident weights) ---
  cvt_f16<<<(szLin + TB - 1) / TB, TB, 0, stream>>>(lin_w, lin16, szLin);
  cvt_f16<<<(szLin + TB - 1) / TB, TB, 0, stream>>>(gatw, gat16, szLin);
  cvt_f16<<<(szWih + TB - 1) / TB, TB, 0, stream>>>(wih, wih16, szWih);
  cvt_f16<<<(szCfw + TB - 1) / TB, TB, 0, stream>>>(cfw, cfw16, szCfw);
  cvt_f16<<<gND, TB, 0, stream>>>(x, hA16, (int)nd);

  // --- degree / dinv ---
  init_deg<<<gN, TB, 0, stream>>>(dinv, N);
  accum_deg<<<gE, TB, 0, stream>>>(ei, dinv, E);
  deg_to_dinv<<<gN, TB, 0, stream>>>(dinv, N);

  // --- gcn: zw = x @ lin_w^T + lin_b ; normalized scatter ; bn1+relu ---
  gemm128_wmma_f16<<<gTile, TB, 0, stream>>>(hA16, lin16, lin_b, zw, N);
  zero_f<<<gND, TB, 0, stream>>>(agg, (int)nd);
  gcn_scatter<<<E + N, 128, 0, stream>>>(ei, dinv, zw, agg, E, N);
  add_bn_relu<<<gND, TB, 0, stream>>>(agg, nullptr, bn1g, bn1b, bn1m, bn1v,
                                      h0, (int)nd);

  // --- rfca #1 (bn2+relu fused; writes f32 h1 + f16 mirror hA16) ---
  rfca_fused<<<gTile, TB, 0, stream>>>(h0, genw, geng, genb, genm, genv,
                                       c1w, c1b, bnmg, bnmb, bnmm, bnmv,
                                       chw, chb, cww, cwb, cfw16, cfb,
                                       bn2g, bn2b, bn2m, bn2v, h1, hA16, N);

  // --- gat ---
  gemm128_wmma_f16<<<gTile, TB, 0, stream>>>(hA16, gat16, nullptr, zw, N);
  gat_coef<<<gN, TB, 0, stream>>>(zw, gasrc, gadst, zs, zd, N);
  zero_f<<<gN, TB, 0, stream>>>((float*)mEnc, N);   // 0u acts as -inf key
  zero_f<<<gN, TB, 0, stream>>>(sSum, N);
  gat_max<<<gEN, TB, 0, stream>>>(ei, zs, zd, mEnc, E, N);
  gat_sum<<<gEN, TB, 0, stream>>>(ei, zs, zd, mEnc, sSum, E, N);
  zero_f<<<gND, TB, 0, stream>>>(agg, (int)nd);
  gat_scatter<<<E + N, 128, 0, stream>>>(ei, zs, zd, mEnc, sSum, zw, agg, E, N);
  add_bn_relu<<<gND, TB, 0, stream>>>(agg, gatb, bn2g, bn2b, bn2m, bn2v,
                                      h0, (int)nd);

  // --- rfca #2 ---
  rfca_fused<<<gTile, TB, 0, stream>>>(h0, genw, geng, genb, genm, genv,
                                       c1w, c1b, bnmg, bnmb, bnmm, bnmv,
                                       chw, chb, cww, cwb, cfw16, cfb,
                                       bn2g, bn2b, bn2m, bn2v, h1, hA16, N);

  // --- 4 lstm-gate layers (f16 ping-pong; f32 h kept for the head) ---
  const _Float16* s16[4] = { hA16, hB16, hA16, hB16 };
  _Float16*       d16[4] = { hB16, hA16, hB16, hA16 };
  float*          d32[4] = { h0,   h1,   h0,   h1   };
  for (int l = 0; l < 4; ++l) {
    lstm_layer_wmma<<<gTile, TB, 0, stream>>>(
        s16[l], wih16 + (size_t)l * 512 * DF, bih + (size_t)l * 512,
        bhh + (size_t)l * 512, d32[l], d16[l], N);
  }

  // --- head: relu(128->64), 64->10, log_softmax ---
  head_kernel<<<gN, TB, 0, stream>>>(h1, l1w, l1b, l2w, l2b, out, N);
}
